// MAGNN_lp_12051678233156
// MI455X (gfx1250) — compile-verified, matrix operand order checked by reading, not run on hardware
//
#include <hip/hip_runtime.h>
#include <hip/hip_bf16.h>

typedef unsigned int   u32;
typedef unsigned short u16;

typedef __attribute__((ext_vector_type(16))) __bf16 v16bf;
typedef __attribute__((ext_vector_type(8)))  float  v8f;

union FragB { uint4 u[2]; v16bf v; };

#define NT_C 33333
#define F_C  256
#define H_C  128
#define K_C  1024
#define B_C  32768
#define N_C  (3*NT_C)
#define HPAD 1032   // 1024 + 8 bf16 pad -> 2064B row stride, bank-spread

// ---------------- helpers ----------------
__device__ inline u16 f2bf(float x) {
    u32 u = __float_as_uint(x);
    u32 r = (u + 0x7FFFu + ((u >> 16) & 1u)) >> 16;
    return (u16)r;
}
__device__ inline float eluf(float x) { return x > 0.f ? x : expm1f(x); }

// ---------------- elementwise / setup kernels ----------------
__global__ void zero_f32_kernel(float* p, int n) {
    int i = blockIdx.x * blockDim.x + threadIdx.x;
    if (i < n) p[i] = 0.f;
}

__global__ void f32_to_bf16_kernel(const float* __restrict__ in, u16* __restrict__ out, int n) {
    int i = blockIdx.x * blockDim.x + threadIdx.x;
    if (i < n) out[i] = f2bf(in[i]);
}

// Pack f32 weight [K, Nout] (row-major) into per-lane WMMA B-fragment layout:
// u32 index = ((kt*(Nout/16)+nt)*32 + lane)*8 + v ; pair = {W[k,col], W[k+1,col]}
__global__ void pack_b_kernel(const float* __restrict__ W, u32* __restrict__ out, int K, int Nout) {
    int total = (K >> 1) * Nout;
    int tid = blockIdx.x * blockDim.x + threadIdx.x;
    if (tid >= total) return;
    int v    = tid & 7;
    int lane = (tid >> 3) & 31;
    int rest = tid >> 8;
    int ntiles = Nout >> 4;
    int nt = rest % ntiles;
    int kt = rest / ntiles;
    int col = (nt << 4) + (lane & 15);
    int k   = (kt << 5) + ((lane >> 4) << 3) + ((v & 3) << 1) + ((v >> 2) << 4);
    u32 lo = f2bf(W[(size_t)k * Nout + col]);
    u32 hi = f2bf(W[(size_t)(k + 1) * Nout + col]);
    out[tid] = lo | (hi << 16);
}

// ---------------- generic bf16 WMMA GEMM ----------------
// out[M,Nout] = act(A[M,K](bf16) @ Bpack + bias). 8 waves/block cover 128 cols.
__global__ void gemm_bf16_kernel(const u16* __restrict__ A, const u32* __restrict__ Bp,
                                 const float* __restrict__ bias, const int* __restrict__ rowMap,
                                 float* __restrict__ outF, u16* __restrict__ outB,
                                 int M, int K, int Nout, int act) {
    int lane = threadIdx.x & 31;
    int wave = threadIdx.x >> 5;
    int nt   = blockIdx.y * 8 + wave;
    if ((nt << 4) >= Nout) return;
    int mt   = blockIdx.x;
    int rowA = (mt << 4) + (lane & 15);
    if (rowA >= M) rowA = M - 1;                 // clamp: keep EXEC full
    int klo  = (lane >> 4) << 3;
    int ntiles = Nout >> 4;
    v8f acc = {};
    for (int kt = 0; kt < (K >> 5); ++kt) {
        FragB a, b;
        const u16* ap = A + (size_t)rowA * K + (kt << 5) + klo;
        a.u[0] = *(const uint4*)ap;
        a.u[1] = *(const uint4*)(ap + 16);
        const uint4* bp = (const uint4*)(Bp + ((((size_t)kt * ntiles + nt) << 5) + lane) * 8);
        b.u[0] = bp[0]; b.u[1] = bp[1];
        acc = __builtin_amdgcn_wmma_f32_16x16x32_bf16(false, a.v, false, b.v, (short)0, acc, false, false);
    }
    int col   = (nt << 4) + (lane & 15);
    float bv  = bias ? bias[col] : 0.f;
    int rbase = (mt << 4) + ((lane >> 4) << 3);
#pragma unroll
    for (int v = 0; v < 8; ++v) {
        int gr = rbase + v;
        if (gr >= M) break;
        float x = acc[v] + bv;
        if (act == 1) x = eluf(x);
        int orow = rowMap ? rowMap[gr] : gr;
        if (outF) outF[(size_t)orow * Nout + col] = x;
        if (outB) outB[(size_t)orow * Nout + col] = f2bf(x);
    }
}

// ---------------- gather + RotatE0 rotation + mean over L ----------------
__global__ void gather_rotate_kernel(const float* __restrict__ features, const int* __restrict__ idx,
                                     const int* __restrict__ etypes, const float* __restrict__ r_vec,
                                     u16* __restrict__ outB, int PB, int Bsz, int L) {
    int lane = threadIdx.x & 31;
    int wid  = blockIdx.x * 8 + (threadIdx.x >> 5);
    if (wid >= PB) return;
    int p = wid / Bsz;
    float4 acc = {0.f, 0.f, 0.f, 0.f};
    for (int t = 0; t < L; ++t) {
        int node = idx[(size_t)wid * L + t];
        float4 f = ((const float4*)(features + (size_t)node * H_C))[lane];
        if (etypes && t < L - 1) {
            int et = etypes[p * (L - 1) + t] % 3;
            const float* rp = r_vec + ((size_t)et * 64 + lane * 2) * 2;
            float ar = rp[0], ai = rp[1], br = rp[2], bi = rp[3];
            float n0 = rsqrtf(ar * ar + ai * ai); ar *= n0; ai *= n0;
            float n1 = rsqrtf(br * br + bi * bi); br *= n1; bi *= n1;
            float re0 = f.x * ar - f.y * ai, im0 = f.x * ai + f.y * ar;
            float re1 = f.z * br - f.w * bi, im1 = f.z * bi + f.w * br;
            f.x = re0; f.y = im0; f.z = re1; f.w = im1;
        }
        acc.x += f.x; acc.y += f.y; acc.z += f.z; acc.w += f.w;
    }
    float inv = 1.f / (float)L;
    u32 lo = (u32)f2bf(acc.x * inv) | ((u32)f2bf(acc.y * inv) << 16);
    u32 hi = (u32)f2bf(acc.z * inv) | ((u32)f2bf(acc.w * inv) << 16);
    uint2 pk; pk.x = lo; pk.y = hi;
    ((uint2*)(outB + (size_t)wid * H_C))[lane] = pk;
}

// ---------------- attention-score pass (no h materialization) ----------------
// Per block: 16 batch rows, one metapath p. Compute h tile [16x1024] (WMMA), elu,
// stage bf16 in LDS, re-feed as A for h@a1 (WMMA), tanh*a2, reduce -> atomic s_acc[p].
__global__ void attn_score_kernel(const u16* __restrict__ meta, const u32* __restrict__ Wp,
                                  const float* __restrict__ bW, const u32* __restrict__ a1p,
                                  const float* __restrict__ ba1, const float* __restrict__ a2,
                                  float* __restrict__ s_acc, int Bsz) {
    __shared__ u16 hT[16 * HPAD];
    int lane = threadIdx.x & 31;
    int wave = threadIdx.x >> 5;
    int p = blockIdx.y;
    int rowA = p * Bsz + (blockIdx.x << 4) + (lane & 15);
    int klo  = (lane >> 4) << 3;

    v8f acc[8] = {};
    for (int kt = 0; kt < 4; ++kt) {                      // K = 128 (meta)
        FragB a;
        const u16* ap = meta + (size_t)rowA * H_C + (kt << 5) + klo;
        a.u[0] = *(const uint4*)ap;
        a.u[1] = *(const uint4*)(ap + 16);
#pragma unroll
        for (int n2 = 0; n2 < 8; ++n2) {
            int nt = wave * 8 + n2;                       // Nout = 1024
            FragB b;
            const uint4* bp = (const uint4*)(Wp + ((((size_t)kt * 64 + nt) << 5) + lane) * 8);
            b.u[0] = bp[0]; b.u[1] = bp[1];
            acc[n2] = __builtin_amdgcn_wmma_f32_16x16x32_bf16(false, a.v, false, b.v, (short)0, acc[n2], false, false);
        }
    }
    int rl = (lane >> 4) << 3;
#pragma unroll
    for (int n2 = 0; n2 < 8; ++n2) {
        int col = wave * 128 + n2 * 16 + (lane & 15);
        float bw = bW[col];
#pragma unroll
        for (int v = 0; v < 8; ++v) {
            float x = eluf(acc[n2][v] + bw);
            hT[(rl + v) * HPAD + col] = f2bf(x);
        }
    }
    __syncthreads();

    v8f acc2 = {};
    for (int kt = 0; kt < 32; ++kt) {                     // K = 1024 (h)
        FragB a, b;
        const u16* ap = &hT[(lane & 15) * HPAD + (kt << 5) + klo];
        a.u[0] = *(const uint4*)ap;
        a.u[1] = *(const uint4*)(ap + 16);
        const uint4* bp = (const uint4*)(a1p + ((((size_t)kt * 8 + wave) << 5) + lane) * 8);
        b.u[0] = bp[0]; b.u[1] = bp[1];
        acc2 = __builtin_amdgcn_wmma_f32_16x16x32_bf16(false, a.v, false, b.v, (short)0, acc2, false, false);
    }
    int col = wave * 16 + (lane & 15);
    float ba = ba1[col], aa = a2[col];
    float s = 0.f;
#pragma unroll
    for (int v = 0; v < 8; ++v) s += tanhf(acc2[v] + ba) * aa;
    for (int m = 16; m >= 1; m >>= 1) s += __shfl_xor(s, m, 32);
    if (lane == 0) atomicAdd(&s_acc[p], s);
}

__global__ void softmax_w_kernel(const float* __restrict__ s_acc, float* __restrict__ wdev, float invB) {
    if (threadIdx.x == 0 && blockIdx.x == 0) {
        const int off[3] = {0, 3, 6}, cnt[3] = {3, 3, 4};
        for (int gi = 0; gi < 3; ++gi) {
            float m = -1e30f;
            for (int i = 0; i < cnt[gi]; ++i) { float v = s_acc[off[gi] + i] * invB; if (v > m) m = v; }
            float e[4], sum = 0.f;
            for (int i = 0; i < cnt[gi]; ++i) { e[i] = expf(s_acc[off[gi] + i] * invB - m); sum += e[i]; }
            for (int i = 0; i < cnt[gi]; ++i) wdev[off[gi] + i] = e[i] / sum;
        }
    }
}

// ---------------- weighted combine pass: hu = sum_p w[p]*elu(meta_p @ W + bW) ----------------
__global__ void weighted_combine_kernel(const u16* __restrict__ meta, const u32* __restrict__ Wp,
                                        const float* __restrict__ bW, const float* __restrict__ wg,
                                        u16* __restrict__ outB, int P, int Bsz) {
    int lane = threadIdx.x & 31;
    int wave = threadIdx.x >> 5;
    int nt   = blockIdx.y * 8 + wave;                     // Nout = 1024
    int klo  = (lane >> 4) << 3;
    int rloc = (blockIdx.x << 4) + (lane & 15);
    int col  = (nt << 4) + (lane & 15);
    float bw = bW[col];
    v8f tot = {};
    for (int p = 0; p < P; ++p) {
        float wp = wg[p];
        v8f acc = {};
        int rowA = p * Bsz + rloc;
        for (int kt = 0; kt < 4; ++kt) {
            FragB a, b;
            const u16* ap = meta + (size_t)rowA * H_C + (kt << 5) + klo;
            a.u[0] = *(const uint4*)ap;
            a.u[1] = *(const uint4*)(ap + 16);
            const uint4* bp = (const uint4*)(Wp + ((((size_t)kt * 64 + nt) << 5) + lane) * 8);
            b.u[0] = bp[0]; b.u[1] = bp[1];
            acc = __builtin_amdgcn_wmma_f32_16x16x32_bf16(false, a.v, false, b.v, (short)0, acc, false, false);
        }
#pragma unroll
        for (int v = 0; v < 8; ++v) tot[v] += wp * eluf(acc[v] + bw);
    }
    int rbase = (blockIdx.x << 4) + ((lane >> 4) << 3);
#pragma unroll
    for (int v = 0; v < 8; ++v)
        outB[(size_t)(rbase + v) * K_C + col] = f2bf(tot[v]);
}

// ---------------- LayerNorm(128) + ELU, one wave per row ----------------
__global__ void ln_elu_kernel(const float* __restrict__ X, const float* __restrict__ g,
                              const float* __restrict__ be, float* __restrict__ outF, int rows) {
    int lane = threadIdx.x & 31;
    int row  = blockIdx.x * 8 + (threadIdx.x >> 5);
    if (row >= rows) return;
    float4 x = ((const float4*)(X + (size_t)row * H_C))[lane];
    float s = x.x + x.y + x.z + x.w;
    float q = x.x * x.x + x.y * x.y + x.z * x.z + x.w * x.w;
    for (int m = 16; m >= 1; m >>= 1) { s += __shfl_xor(s, m, 32); q += __shfl_xor(q, m, 32); }
    float mean = s * (1.f / 128.f);
    float var  = q * (1.f / 128.f) - mean * mean;
    float inv  = rsqrtf(var + 1e-5f);
    float4 gg = ((const float4*)g)[lane];
    float4 bb = ((const float4*)be)[lane];
    float4 y;
    y.x = eluf((x.x - mean) * inv * gg.x + bb.x);
    y.y = eluf((x.y - mean) * inv * gg.y + bb.y);
    y.z = eluf((x.z - mean) * inv * gg.z + bb.z);
    y.w = eluf((x.w - mean) * inv * gg.w + bb.w);
    ((float4*)(outF + (size_t)row * H_C))[lane] = y;
}

__global__ void mul_bf16_kernel(const float* __restrict__ a, const float* __restrict__ b,
                                u16* __restrict__ out, int n) {
    int i = blockIdx.x * blockDim.x + threadIdx.x;
    if (i < n) out[i] = f2bf(a[i] * b[i]);
}

__global__ void combine_z_kernel(const float* __restrict__ xij, const float* __restrict__ el,
                                 const float* __restrict__ beta, u16* __restrict__ out, int n) {
    int i = blockIdx.x * blockDim.x + threadIdx.x;
    if (i < n) out[i] = f2bf(xij[i] + el[i] * beta[0]);
}

__global__ void final_head_kernel(const float* __restrict__ Y, const float* __restrict__ w2,
                                  const float* __restrict__ b2, float* __restrict__ out, int rows) {
    int lane = threadIdx.x & 31;
    int row  = blockIdx.x * 8 + (threadIdx.x >> 5);
    if (row >= rows) return;
    float4 y = ((const float4*)(Y + (size_t)row * H_C))[lane];
    float4 w = ((const float4*)w2)[lane];
    float s = y.x * w.x + y.y * w.y + y.z * w.z + y.w * w.w;
    for (int m = 16; m >= 1; m >>= 1) s += __shfl_xor(s, m, 32);
    if (lane == 0) out[row] = 1.f / (1.f + expf(-(s + b2[0])));
}

// ======================= host side =======================
extern "C" void kernel_launch(void* const* d_in, const int* in_sizes, int n_in,
                              void* d_out, int out_size, void* d_ws, size_t ws_size,
                              hipStream_t stream) {
    const float* feats[3] = {(const float*)d_in[0], (const float*)d_in[1], (const float*)d_in[2]};
    const int*   node_idx   = (const int*)d_in[3];
    const int*   user_idx   = (const int*)d_in[4];
    const int*   item_idx   = (const int*)d_in[5];
    const int*   link_idx   = (const int*)d_in[6];
    const int*   et_user    = (const int*)d_in[7];
    const int*   et_item    = (const int*)d_in[8];
    const float* fc_w       = (const float*)d_in[9];
    const float* fc_b       = (const float*)d_in[10];
    const float* r_vec      = (const float*)d_in[11];
    const float* W_layers   = (const float*)d_in[12];
    const float* bW_layers  = (const float*)d_in[13];
    const float* a1_layers  = (const float*)d_in[14];
    const float* ba1_layers = (const float*)d_in[15];
    const float* a2_layers  = (const float*)d_in[16];
    const float* wu_w = (const float*)d_in[17]; const float* wu_b = (const float*)d_in[18];
    const float* wi_w = (const float*)d_in[19]; const float* wi_b = (const float*)d_in[20];
    const float* xij_w = (const float*)d_in[21]; const float* xij_b = (const float*)d_in[22];
    const float* xij_g = (const float*)d_in[23]; const float* xij_be = (const float*)d_in[24];
    const float* link_w = (const float*)d_in[25]; const float* link_b = (const float*)d_in[26];
    const float* link_g = (const float*)d_in[27]; const float* link_be = (const float*)d_in[28];
    const float* beta   = (const float*)d_in[29];
    const float* lin1_w = (const float*)d_in[30]; const float* lin1_b = (const float*)d_in[31];
    const float* lin_g  = (const float*)d_in[32]; const float* lin_be = (const float*)d_in[33];
    const float* lin2_w = (const float*)d_in[34]; const float* lin2_b = (const float*)d_in[35];
    float* out = (float*)d_out;

    char* ws = (char*)d_ws;
    size_t off = 0;
    auto carve = [&](size_t bytes) -> void* {
        void* p = ws + off;
        off = (off + bytes + 255) & ~(size_t)255;
        return p;
    };
    u16* featsB  = (u16*)carve((size_t)3 * NT_C * F_C * 2);
    u32* fcP     = (u32*)carve((size_t)3 * (F_C * H_C / 2) * 4);
    u32* WPk     = (u32*)carve((size_t)3 * (H_C * K_C / 2) * 4);
    u32* a1P     = (u32*)carve((size_t)3 * (K_C * H_C / 2) * 4);
    u32* wuP     = (u32*)carve((size_t)(K_C * H_C / 2) * 4);
    u32* wiP     = (u32*)carve((size_t)(K_C * H_C / 2) * 4);
    u32* linkP   = (u32*)carve((size_t)(K_C * H_C / 2) * 4);
    u32* xijP    = (u32*)carve((size_t)(H_C * H_C / 2) * 4);
    u32* lin1P   = (u32*)carve((size_t)(H_C * H_C / 2) * 4);
    float* features = (float*)carve((size_t)N_C * H_C * 4);
    u16* metaU   = (u16*)carve((size_t)3 * B_C * H_C * 2);
    u16* metaI   = (u16*)carve((size_t)3 * B_C * H_C * 2);
    u16* metaL   = (u16*)carve((size_t)4 * B_C * H_C * 2);
    float* sacc  = (float*)carve(64);
    float* wdev  = (float*)carve(64);
    u16* hu      = (u16*)carve((size_t)B_C * K_C * 2);
    u16* hi      = (u16*)carve((size_t)B_C * K_C * 2);
    u16* hl      = (u16*)carve((size_t)B_C * K_C * 2);
    float* huP   = (float*)carve((size_t)B_C * H_C * 4);
    float* hiP   = (float*)carve((size_t)B_C * H_C * 4);
    float* hlLin = (float*)carve((size_t)B_C * H_C * 4);
    float* elAct = (float*)carve((size_t)B_C * H_C * 4);
    u16* prodB   = (u16*)carve((size_t)B_C * H_C * 2);
    float* xijLin = (float*)carve((size_t)B_C * H_C * 4);
    float* xijAct = (float*)carve((size_t)B_C * H_C * 4);
    u16* zB      = (u16*)carve((size_t)B_C * H_C * 2);
    float* yLin  = (float*)carve((size_t)B_C * H_C * 4);
    float* yAct  = (float*)carve((size_t)B_C * H_C * 4);
    (void)ws_size; (void)n_in; (void)in_sizes; (void)out_size;

    // 0) zero score accumulators (must be done every launch)
    zero_f32_kernel<<<1, 32, 0, stream>>>(sacc, 10);

    // 1) convert raw features to bf16
    {
        int n = NT_C * F_C;
        int g = (n + 255) / 256;
        for (int t = 0; t < 3; ++t)
            f32_to_bf16_kernel<<<g, 256, 0, stream>>>(feats[t], featsB + (size_t)t * n, n);
    }

    // 2) pack all GEMM weights into WMMA B-fragment layout
    for (int t = 0; t < 3; ++t)
        pack_b_kernel<<<(F_C * H_C / 2 + 255) / 256, 256, 0, stream>>>(fc_w + (size_t)t * F_C * H_C, fcP + (size_t)t * (F_C * H_C / 2), F_C, H_C);
    for (int g2 = 0; g2 < 3; ++g2) {
        pack_b_kernel<<<(H_C * K_C / 2 + 255) / 256, 256, 0, stream>>>(W_layers + (size_t)g2 * H_C * K_C, WPk + (size_t)g2 * (H_C * K_C / 2), H_C, K_C);
        pack_b_kernel<<<(K_C * H_C / 2 + 255) / 256, 256, 0, stream>>>(a1_layers + (size_t)g2 * K_C * H_C, a1P + (size_t)g2 * (K_C * H_C / 2), K_C, H_C);
    }
    pack_b_kernel<<<(K_C * H_C / 2 + 255) / 256, 256, 0, stream>>>(wu_w, wuP, K_C, H_C);
    pack_b_kernel<<<(K_C * H_C / 2 + 255) / 256, 256, 0, stream>>>(wi_w, wiP, K_C, H_C);
    pack_b_kernel<<<(K_C * H_C / 2 + 255) / 256, 256, 0, stream>>>(link_w, linkP, K_C, H_C);
    pack_b_kernel<<<(H_C * H_C / 2 + 255) / 256, 256, 0, stream>>>(xij_w, xijP, H_C, H_C);
    pack_b_kernel<<<(H_C * H_C / 2 + 255) / 256, 256, 0, stream>>>(lin1_w, lin1P, H_C, H_C);

    // 3) feature projection GEMMs, scattered through node_idx
    {
        dim3 grid((NT_C + 15) / 16, 1);
        for (int t = 0; t < 3; ++t)
            gemm_bf16_kernel<<<grid, 256, 0, stream>>>(featsB + (size_t)t * NT_C * F_C,
                fcP + (size_t)t * (F_C * H_C / 2), fc_b + t * H_C, node_idx + t * NT_C,
                features, nullptr, NT_C, F_C, H_C, 0);
    }

    // 4) metapath gather + rotation -> bf16 meta encodings
    gather_rotate_kernel<<<(3 * B_C + 7) / 8, 256, 0, stream>>>(features, user_idx, et_user, r_vec, metaU, 3 * B_C, B_C, 4);
    gather_rotate_kernel<<<(3 * B_C + 7) / 8, 256, 0, stream>>>(features, item_idx, et_item, r_vec, metaI, 3 * B_C, B_C, 4);
    gather_rotate_kernel<<<(4 * B_C + 7) / 8, 256, 0, stream>>>(features, link_idx, nullptr, r_vec, metaL, 4 * B_C, B_C, 2);

    // 5) attention score pass (h never materialized)
    attn_score_kernel<<<dim3(B_C / 16, 3), 256, 0, stream>>>(metaU, WPk, bW_layers, a1P, ba1_layers, a2_layers, sacc, B_C);
    attn_score_kernel<<<dim3(B_C / 16, 3), 256, 0, stream>>>(metaI, WPk + (size_t)(H_C * K_C / 2), bW_layers + K_C,
        a1P + (size_t)(K_C * H_C / 2), ba1_layers + H_C, a2_layers + H_C, sacc + 3, B_C);
    attn_score_kernel<<<dim3(B_C / 16, 4), 256, 0, stream>>>(metaL, WPk + (size_t)2 * (H_C * K_C / 2), bW_layers + 2 * K_C,
        a1P + (size_t)2 * (K_C * H_C / 2), ba1_layers + 2 * H_C, a2_layers + 2 * H_C, sacc + 6, B_C);

    softmax_w_kernel<<<1, 32, 0, stream>>>(sacc, wdev, 1.f / (float)B_C);

    // 6) weighted combine (recompute h, fold softmax weights in-register)
    weighted_combine_kernel<<<dim3(B_C / 16, 8), 256, 0, stream>>>(metaU, WPk, bW_layers, wdev, hu, 3, B_C);
    weighted_combine_kernel<<<dim3(B_C / 16, 8), 256, 0, stream>>>(metaI, WPk + (size_t)(H_C * K_C / 2), bW_layers + K_C, wdev + 3, hi, 3, B_C);
    weighted_combine_kernel<<<dim3(B_C / 16, 8), 256, 0, stream>>>(metaL, WPk + (size_t)2 * (H_C * K_C / 2), bW_layers + 2 * K_C, wdev + 6, hl, 4, B_C);

    // 7) head projections
    dim3 g128(B_C / 16, 1);
    gemm_bf16_kernel<<<g128, 256, 0, stream>>>(hu, wuP, wu_b, nullptr, huP, nullptr, B_C, K_C, H_C, 0);
    gemm_bf16_kernel<<<g128, 256, 0, stream>>>(hi, wiP, wi_b, nullptr, hiP, nullptr, B_C, K_C, H_C, 0);
    gemm_bf16_kernel<<<g128, 256, 0, stream>>>(hl, linkP, link_b, nullptr, hlLin, nullptr, B_C, K_C, H_C, 0);

    ln_elu_kernel<<<(B_C + 7) / 8, 256, 0, stream>>>(hlLin, link_g, link_be, elAct, B_C);

    mul_bf16_kernel<<<(B_C * H_C + 255) / 256, 256, 0, stream>>>(huP, hiP, prodB, B_C * H_C);
    gemm_bf16_kernel<<<g128, 256, 0, stream>>>(prodB, xijP, xij_b, nullptr, xijLin, nullptr, B_C, H_C, H_C, 0);
    ln_elu_kernel<<<(B_C + 7) / 8, 256, 0, stream>>>(xijLin, xij_g, xij_be, xijAct, B_C);

    combine_z_kernel<<<(B_C * H_C + 255) / 256, 256, 0, stream>>>(xijAct, elAct, beta, zB, B_C * H_C);
    gemm_bf16_kernel<<<g128, 256, 0, stream>>>(zB, lin1P, lin1_b, nullptr, yLin, nullptr, B_C, H_C, H_C, 0);
    ln_elu_kernel<<<(B_C + 7) / 8, 256, 0, stream>>>(yLin, lin_g, lin_be, yAct, B_C);

    final_head_kernel<<<(B_C + 7) / 8, 256, 0, stream>>>(yAct, lin2_w, lin2_b, out, B_C);
}
